// GraphAtten_64269890617452
// MI455X (gfx1250) — matrix operand order repaired and checked by reading, not compiled, not run on hardware
//
#include <hip/hip_runtime.h>
#include <hip/hip_bf16.h>

typedef __attribute__((ext_vector_type(16))) __bf16 v16bf;
typedef __attribute__((ext_vector_type(8)))  float  v8f;

#define ALPHA_LRELU 0.2f
#define BN_EPS 1e-5f

// ---------- WMMA fragment packing (per CDNA5 ISA 7.12.2) ----------
// A (16x32 bf16): low lanes hold K 0..7 / 16..23, high lanes K 8..15 / 24..31
__device__ __forceinline__ unsigned apack_idx(int m, int k, int ktiles) {
  int mt = m >> 4, kt = k >> 5;
  int mi = m & 15, kk = k & 31;
  int g  = kk & 15;
  int lane = mi + ((g >> 3) << 4);
  int j = (g & 7) + ((kk >> 4) << 3);
  return (((unsigned)(mt * ktiles + kt)) << 9) + (unsigned)((lane << 4) + j);
}
// B (32x16 bf16): low lanes hold K 0..7 / 16..23 at column n, high lanes K 8..15 / 24..31
__device__ __forceinline__ unsigned bpack_idx(int k, int n, int ntiles) {
  int kt = k >> 5, nt = n >> 4;
  int ki = k & 31, ni = n & 15;
  int lane = ni + (((ki >> 3) & 1) << 4);
  int j = (ki & 7) + ((ki >> 4) << 3);
  return (((unsigned)(kt * ntiles + nt)) << 9) + (unsigned)((lane << 4) + j);
}

__device__ __forceinline__ v8f wmma_bf16(v16bf a, v16bf b, v8f c) {
  return __builtin_amdgcn_wmma_f32_16x16x32_bf16(false, a, false, b, (short)0, c,
                                                 false, false);
}

// async global -> LDS copy, 16B per lane, tracked by ASYNCcnt
__device__ __forceinline__ void async_copy_b128(unsigned lds_off, const void* gaddr) {
  asm volatile("global_load_async_to_lds_b128 %0, %1, off"
               :: "v"(lds_off), "v"(gaddr) : "memory");
}
__device__ __forceinline__ void wait_async0() {
  asm volatile("s_wait_asynccnt 0x0" ::: "memory");
}

// ---------- generic wave-level 32x64 GEMM on packed operands ----------
template <int KTILES, int BNTILES>
__device__ __forceinline__ void wave_gemm_32x64(const __bf16* __restrict__ Ap,
                                                const __bf16* __restrict__ Bp,
                                                int mt0, int nt0, int lane,
                                                v8f acc[2][4]) {
  unsigned a0 = (((unsigned)(mt0 + 0) * KTILES) << 9) + (unsigned)(lane << 4);
  unsigned a1 = (((unsigned)(mt0 + 1) * KTILES) << 9) + (unsigned)(lane << 4);
  unsigned b[4];
#pragma unroll
  for (int j = 0; j < 4; ++j) b[j] = (((unsigned)(nt0 + j)) << 9) + (unsigned)(lane << 4);
  for (int kt = 0; kt < KTILES; ++kt) {
    v16bf av0 = *(const v16bf*)(Ap + a0); a0 += 512;
    v16bf av1 = *(const v16bf*)(Ap + a1); a1 += 512;
#pragma unroll
    for (int j = 0; j < 4; ++j) {
      v16bf bv = *(const v16bf*)(Bp + b[j]); b[j] += BNTILES * 512;
      acc[0][j] = wmma_bf16(av0, bv, acc[0][j]);
      acc[1][j] = wmma_bf16(av1, bv, acc[1][j]);
    }
  }
}

// ---------- conversion / packing kernels ----------
__global__ __launch_bounds__(256) void k_cvt_x(const float* __restrict__ x,
                                               __bf16* __restrict__ xA,
                                               __bf16* __restrict__ hA) {
  int idx = blockIdx.x * 256 + threadIdx.x;   // 4096*512
  int row = idx >> 9, col = idx & 511;
  __bf16 b = (__bf16)x[idx];
  xA[apack_idx(row, col, 16)] = b;            // QKV input, K=512
  hA[apack_idx(row, col, 20)] = b;            // concat matrix, K=640, cols 0..511
}

__global__ __launch_bounds__(256) void k_pack_b(const float* __restrict__ src,
                                                __bf16* __restrict__ dst,
                                                int K, int N, int total) {
  int idx = blockIdx.x * 256 + threadIdx.x;
  if (idx >= total) return;
  int kn = K * N;
  int mat = idx / kn, rem = idx - mat * kn;
  int k = rem / N, n = rem - k * N;
  dst[(size_t)mat * kn + bpack_idx(k, n, N >> 4)] = (__bf16)src[idx];
}

// ---------- QKV projection: x[4096,512] @ W[512,128] per (which,head) ----------
__global__ __launch_bounds__(256) void k_qkv(const __bf16* __restrict__ xA,
                                             const __bf16* __restrict__ WqB,
                                             const __bf16* __restrict__ WkB,
                                             const __bf16* __restrict__ WvB,
                                             const float* __restrict__ bq,
                                             const float* __restrict__ bk,
                                             const float* __restrict__ bvp,
                                             __bf16* __restrict__ qA,
                                             __bf16* __restrict__ kB,
                                             __bf16* __restrict__ vB) {
  int lane = threadIdx.x & 31, wid = threadIdx.x >> 5;
  int mt0 = blockIdx.x * 8 + (wid & 3) * 2;
  int nt0 = (wid >> 2) * 4;
  int which = blockIdx.y >> 3, head = blockIdx.y & 7;
  const __bf16* Bp = (which == 0 ? WqB : which == 1 ? WkB : WvB) + head * (512 * 128);
  const float* bias = (which == 0 ? bq : which == 1 ? bk : bvp) + head * 128;
  v8f acc[2][4] = {};
  wave_gemm_32x64<16, 8>(xA, Bp, mt0, nt0, lane, acc);
  int rbase = ((lane >> 4) << 3), cl = lane & 15;
  __bf16* qh = qA + head * (4096 * 128);
  __bf16* kh = kB + head * (128 * 4096);
  __bf16* vh = vB + head * (4096 * 128);
#pragma unroll
  for (int i = 0; i < 2; ++i)
#pragma unroll
    for (int j = 0; j < 4; ++j) {
      int d = (nt0 + j) * 16 + cl;
      float bb = bias[d];
#pragma unroll
      for (int r = 0; r < 8; ++r) {
        int row = (mt0 + i) * 16 + rbase + r;
        float z = acc[i][j][r] + bb;
        z = z > 0.f ? z : 0.f;
        __bf16 zb = (__bf16)z;
        if (which == 0)      qh[apack_idx(row, d, 4)]      = zb;  // Q as A-frags (K=128)
        else if (which == 1) kh[bpack_idx(d, row, 256)]    = zb;  // K^T as B-frags (128 x 4096)
        else                 vh[bpack_idx(row, d, 8)]      = zb;  // V as B-frags (4096 x 128)
      }
    }
}

// ---------- fused flash attention (leaky-relu + online softmax) ----------
// Double-buffered async global->LDS staging of K^T/V blocks (ASYNCcnt),
// per-wave LDS transpose of P from C-frag layout to A-frag layout (DScnt).
__global__ __launch_bounds__(256) void k_attn(const __bf16* __restrict__ qA,
                                              const __bf16* __restrict__ kB,
                                              const __bf16* __restrict__ vB,
                                              __bf16* __restrict__ mA) {
  __shared__ alignas(16) __bf16 sK[2][4096];   // 8 B-tiles of K^T per buffer
  __shared__ alignas(16) __bf16 sV[2][4096];   // 8 B-tiles of V per buffer
  __shared__ alignas(16) __bf16 sP[8][512];    // per-wave P transpose buffer
  int tid = threadIdx.x;
  int lane = tid & 31, wid = tid >> 5;
  int gw = blockIdx.x * 8 + wid;               // 2048 waves total
  int head = gw >> 8;                          // uniform within a block
  int mt = gw & 255;                           // 16-query-row tile
  const __bf16* Qh = qA + head * (4096 * 128);
  const uint4* Kg = (const uint4*)(kB + head * (128 * 4096));
  const uint4* Vg = (const uint4*)(vB + head * (4096 * 128));

  v16bf qf[4];
#pragma unroll
  for (int dt = 0; dt < 4; ++dt)
    qf[dt] = *(const v16bf*)(Qh + (((unsigned)(mt * 4 + dt)) << 9) + (lane << 4));

  v8f o[8] = {};
  float mx[8], l[8];
#pragma unroll
  for (int r = 0; r < 8; ++r) { mx[r] = -1e30f; l[r] = 0.f; }
  int rbase = ((lane >> 4) << 3), cl = lane & 15;
  __bf16* L = sP[wid];

  // per-thread staging slots: 2 uint4 of K + 2 uint4 of V per key block
  int u4a = tid, u4b = tid + 256;
  int tileA = u4a >> 6, wA = u4a & 63;
  int tileB = u4b >> 6, wB = u4b & 63;
  unsigned kdstA[2], kdstB[2], vdstA[2], vdstB[2];
#pragma unroll
  for (int b = 0; b < 2; ++b) {
    kdstA[b] = (unsigned)(size_t)(const void*)&sK[b][0] + (unsigned)(u4a * 16);
    kdstB[b] = (unsigned)(size_t)(const void*)&sK[b][0] + (unsigned)(u4b * 16);
    vdstA[b] = (unsigned)(size_t)(const void*)&sV[b][0] + (unsigned)(u4a * 16);
    vdstB[b] = (unsigned)(size_t)(const void*)&sV[b][0] + (unsigned)(u4b * 16);
  }
  auto stage = [&](int t, int b) {
    async_copy_b128(kdstA[b], Kg + ((tileA >> 1) * 256 + t * 2 + (tileA & 1)) * 64 + wA);
    async_copy_b128(kdstB[b], Kg + ((tileB >> 1) * 256 + t * 2 + (tileB & 1)) * 64 + wB);
    async_copy_b128(vdstA[b], Vg + (t * 8 + tileA) * 64 + wA);
    async_copy_b128(vdstB[b], Vg + (t * 8 + tileB) * 64 + wB);
  };

  stage(0, 0);
  for (int t = 0; t < 128; ++t) {              // 32 keys per block
    int b = t & 1;
    wait_async0();                             // our async writes for buffer b done
    __syncthreads();                           // everyone's writes visible; prev reads done
    if (t + 1 < 128) stage(t + 1, b ^ 1);      // prefetch next block into other buffer
    const __bf16* cK = sK[b];
    const __bf16* cV = sV[b];

    // S = Q * K^T  (16 x 32)
    v8f s0 = {}, s1 = {};
#pragma unroll
    for (int dt = 0; dt < 4; ++dt) {
      v16bf b0 = *(const v16bf*)(cK + (dt * 2 + 0) * 512 + (lane << 4));
      v16bf b1 = *(const v16bf*)(cK + (dt * 2 + 1) * 512 + (lane << 4));
      s0 = wmma_bf16(qf[dt], b0, s0);
      s1 = wmma_bf16(qf[dt], b1, s1);
    }
    // leaky relu + row max
    float bmax[8];
#pragma unroll
    for (int r = 0; r < 8; ++r) {
      float a = s0[r]; a = a > 0.f ? a : ALPHA_LRELU * a; s0[r] = a;
      float c = s1[r]; c = c > 0.f ? c : ALPHA_LRELU * c; s1[r] = c;
      bmax[r] = fmaxf(a, c);
    }
#pragma unroll
    for (int d = 1; d < 16; d <<= 1)
#pragma unroll
      for (int r = 0; r < 8; ++r) bmax[r] = fmaxf(bmax[r], __shfl_xor(bmax[r], d, 32));
    float sc[8], rs[8];
#pragma unroll
    for (int r = 0; r < 8; ++r) {
      float nm = fmaxf(mx[r], bmax[r]);
      sc[r] = __expf(mx[r] - nm);
      mx[r] = nm;
      float e0 = __expf(s0[r] - nm), e1 = __expf(s1[r] - nm);
      s0[r] = e0; s1[r] = e1;
      rs[r] = e0 + e1;
    }
#pragma unroll
    for (int d = 1; d < 16; d <<= 1)
#pragma unroll
      for (int r = 0; r < 8; ++r) rs[r] += __shfl_xor(rs[r], d, 32);
#pragma unroll
    for (int r = 0; r < 8; ++r) l[r] = l[r] * sc[r] + rs[r];
#pragma unroll
    for (int nt = 0; nt < 8; ++nt)
#pragma unroll
      for (int r = 0; r < 8; ++r) o[nt][r] *= sc[r];

    // transpose P (C-frag layout) -> A-frag layout through per-wave LDS
#pragma unroll
    for (int r = 0; r < 8; ++r) {
      L[(rbase + r) * 32 + cl]      = (__bf16)s0[r];
      L[(rbase + r) * 32 + 16 + cl] = (__bf16)s1[r];
    }
    asm volatile("s_wait_dscnt 0x0" ::: "memory");
    union { v16bf v; uint4 q[2]; } pf;
    int Mr = lane & 15, g8 = (lane >> 4) << 3;
    pf.q[0] = *(const uint4*)&L[Mr * 32 + g8];
    pf.q[1] = *(const uint4*)&L[Mr * 32 + 16 + g8];

    // O += P * V
#pragma unroll
    for (int nt = 0; nt < 8; ++nt) {
      v16bf vf = *(const v16bf*)(cV + nt * 512 + (lane << 4));
      o[nt] = wmma_bf16(pf.v, vf, o[nt]);
    }
  }

  float inv[8];
#pragma unroll
  for (int r = 0; r < 8; ++r) inv[r] = 1.0f / l[r];
#pragma unroll
  for (int nt = 0; nt < 8; ++nt) {
    int col = head * 128 + nt * 16 + cl;
#pragma unroll
    for (int r = 0; r < 8; ++r) {
      int row = mt * 16 + rbase + r;
      mA[apack_idx(row, col, 32)] = (__bf16)(o[nt][r] * inv[r]);   // m: [4096,1024]
    }
  }
}

// ---------- m @ Wm + bm, relu -> h cols 512..639 ----------
__global__ __launch_bounds__(256) void k_mproj(const __bf16* __restrict__ mA,
                                               const __bf16* __restrict__ WmB,
                                               const float* __restrict__ bm,
                                               __bf16* __restrict__ hA) {
  int lane = threadIdx.x & 31, wid = threadIdx.x >> 5;
  int mt0 = blockIdx.x * 8 + (wid & 3) * 2;
  int nt0 = (wid >> 2) * 4;
  v8f acc[2][4] = {};
  wave_gemm_32x64<32, 8>(mA, WmB, mt0, nt0, lane, acc);
  int rbase = ((lane >> 4) << 3), cl = lane & 15;
#pragma unroll
  for (int i = 0; i < 2; ++i)
#pragma unroll
    for (int j = 0; j < 4; ++j) {
      int col = (nt0 + j) * 16 + cl;
      float bb = bm[col];
#pragma unroll
      for (int r = 0; r < 8; ++r) {
        int row = (mt0 + i) * 16 + rbase + r;
        float z = acc[i][j][r] + bb;
        z = z > 0.f ? z : 0.f;
        hA[apack_idx(row, 512 + col, 20)] = (__bf16)z;
      }
    }
}

// ---------- h @ W1 + b1 -> Z1 (f32, pre-BN) ----------
__global__ __launch_bounds__(256) void k_l1(const __bf16* __restrict__ hA,
                                            const __bf16* __restrict__ W1B,
                                            const float* __restrict__ b1,
                                            float* __restrict__ Z1) {
  int lane = threadIdx.x & 31, wid = threadIdx.x >> 5;
  int mt0 = blockIdx.x * 8 + (wid & 3) * 2;
  int nt0 = blockIdx.y * 8 + (wid >> 2) * 4;
  v8f acc[2][4] = {};
  wave_gemm_32x64<20, 40>(hA, W1B, mt0, nt0, lane, acc);
  int rbase = ((lane >> 4) << 3), cl = lane & 15;
#pragma unroll
  for (int i = 0; i < 2; ++i)
#pragma unroll
    for (int j = 0; j < 4; ++j) {
      int col = (nt0 + j) * 16 + cl;
      float bb = b1[col];
#pragma unroll
      for (int r = 0; r < 8; ++r) {
        int row = (mt0 + i) * 16 + rbase + r;
        Z1[(size_t)row * 640 + col] = acc[i][j][r] + bb;
      }
    }
}

// ---------- h1 @ W2 + b2 -> Z2 (f32, pre-BN) ----------
__global__ __launch_bounds__(256) void k_l2(const __bf16* __restrict__ h1A,
                                            const __bf16* __restrict__ W2B,
                                            const float* __restrict__ b2,
                                            float* __restrict__ Z2) {
  int lane = threadIdx.x & 31, wid = threadIdx.x >> 5;
  int mt0 = blockIdx.x * 8 + (wid & 3) * 2;
  int nt0 = blockIdx.y * 8 + (wid >> 2) * 4;
  v8f acc[2][4] = {};
  wave_gemm_32x64<20, 16>(h1A, W2B, mt0, nt0, lane, acc);
  int rbase = ((lane >> 4) << 3), cl = lane & 15;
#pragma unroll
  for (int i = 0; i < 2; ++i)
#pragma unroll
    for (int j = 0; j < 4; ++j) {
      int col = (nt0 + j) * 16 + cl;
      float bb = b2[col];
#pragma unroll
      for (int r = 0; r < 8; ++r) {
        int row = (mt0 + i) * 16 + rbase + r;
        Z2[(size_t)row * 256 + col] = acc[i][j][r] + bb;
      }
    }
}

// ---------- BatchNorm column stats (biased var, matches jnp.var) ----------
__global__ __launch_bounds__(256) void k_bn_stats(const float* __restrict__ Z, int N,
                                                  float* __restrict__ mu,
                                                  float* __restrict__ ri) {
  int cl = threadIdx.x & 63;
  int rs = threadIdx.x >> 6;
  int c = blockIdx.x * 64 + cl;
  float s = 0.f, s2 = 0.f;
  for (int r = rs; r < 4096; r += 4) {
    float z = Z[(size_t)r * N + c];
    s += z; s2 += z * z;
  }
  __shared__ float sb[4][64], sb2[4][64];
  sb[rs][cl] = s; sb2[rs][cl] = s2;
  __syncthreads();
  if (rs == 0) {
    s  = sb[0][cl] + sb[1][cl] + sb[2][cl] + sb[3][cl];
    s2 = sb2[0][cl] + sb2[1][cl] + sb2[2][cl] + sb2[3][cl];
    float m = s * (1.f / 4096.f);
    float v = s2 * (1.f / 4096.f) - m * m;
    mu[c] = m;
    ri[c] = rsqrtf(v + BN_EPS);
  }
}

__global__ __launch_bounds__(256) void k_bn_apply_pack(const float* __restrict__ Z,
                                                       const float* __restrict__ mu,
                                                       const float* __restrict__ ri,
                                                       const float* __restrict__ g,
                                                       const float* __restrict__ be,
                                                       int N, int ktiles,
                                                       __bf16* __restrict__ out) {
  int idx = blockIdx.x * 256 + threadIdx.x;
  int row = idx / N, col = idx - row * N;
  float y = g[col] * (Z[idx] - mu[col]) * ri[col] + be[col];
  y = y > 0.f ? y : 0.f;
  out[apack_idx(row, col, ktiles)] = (__bf16)y;
}

__global__ __launch_bounds__(256) void k_bn_apply_out(const float* __restrict__ Z,
                                                      const float* __restrict__ mu,
                                                      const float* __restrict__ ri,
                                                      const float* __restrict__ g,
                                                      const float* __restrict__ be,
                                                      int N, float* __restrict__ out) {
  int idx = blockIdx.x * 256 + threadIdx.x;
  int row = idx / N, col = idx - row * N;
  float y = g[col] * (Z[idx] - mu[col]) * ri[col] + be[col];
  out[idx] = y > 0.f ? y : 0.f;
}

extern "C" void kernel_launch(void* const* d_in, const int* in_sizes, int n_in,
                              void* d_out, int out_size, void* d_ws, size_t ws_size,
                              hipStream_t stream) {
  (void)in_sizes; (void)n_in; (void)out_size; (void)ws_size;
  const float* x   = (const float*)d_in[0];
  const float* Wq  = (const float*)d_in[1];
  const float* bq  = (const float*)d_in[2];
  const float* Wk  = (const float*)d_in[3];
  const float* bk  = (const float*)d_in[4];
  const float* Wv  = (const float*)d_in[5];
  const float* bvv = (const float*)d_in[6];
  const float* Wm  = (const float*)d_in[7];
  const float* bm  = (const float*)d_in[8];
  const float* W1  = (const float*)d_in[9];
  const float* b1  = (const float*)d_in[10];
  const float* g1  = (const float*)d_in[11];
  const float* be1 = (const float*)d_in[12];
  const float* W2  = (const float*)d_in[13];
  const float* b2  = (const float*)d_in[14];
  const float* g2  = (const float*)d_in[15];
  const float* be2 = (const float*)d_in[16];
  float* out = (float*)d_out;

  char* ws = (char*)d_ws;
  size_t off = 0;
  auto alloc = [&](size_t bytes) -> void* {
    void* p = ws + off;
    off = (off + bytes + 255) & ~(size_t)255;
    return p;
  };
  __bf16* xA  = (__bf16*)alloc(4096ull * 512 * 2);
  __bf16* hA  = (__bf16*)alloc(4096ull * 640 * 2);
  __bf16* WqB = (__bf16*)alloc(8ull * 512 * 128 * 2);
  __bf16* WkB = (__bf16*)alloc(8ull * 512 * 128 * 2);
  __bf16* WvB = (__bf16*)alloc(8ull * 512 * 128 * 2);
  __bf16* WmB = (__bf16*)alloc(1024ull * 128 * 2);
  __bf16* W1B = (__bf16*)alloc(640ull * 640 * 2);
  __bf16* W2B = (__bf16*)alloc(640ull * 256 * 2);
  __bf16* qA  = (__bf16*)alloc(8ull * 4096 * 128 * 2);
  __bf16* kB  = (__bf16*)alloc(8ull * 128 * 4096 * 2);
  __bf16* vB  = (__bf16*)alloc(8ull * 4096 * 128 * 2);
  __bf16* mA  = (__bf16*)alloc(4096ull * 1024 * 2);
  float*  Z1  = (float*)alloc(4096ull * 640 * 4);
  __bf16* h1A = (__bf16*)alloc(4096ull * 640 * 2);
  float*  Z2  = (float*)alloc(4096ull * 256 * 4);
  float*  mu1 = (float*)alloc(640 * 4);
  float*  ri1 = (float*)alloc(640 * 4);
  float*  mu2 = (float*)alloc(256 * 4);
  float*  ri2 = (float*)alloc(256 * 4);

  // stage 0: convert + pack
  k_cvt_x<<<(4096 * 512) / 256, 256, 0, stream>>>(x, xA, hA);
  k_pack_b<<<(8 * 512 * 128) / 256, 256, 0, stream>>>(Wq, WqB, 512, 128, 8 * 512 * 128);
  k_pack_b<<<(8 * 512 * 128) / 256, 256, 0, stream>>>(Wk, WkB, 512, 128, 8 * 512 * 128);
  k_pack_b<<<(8 * 512 * 128) / 256, 256, 0, stream>>>(Wv, WvB, 512, 128, 8 * 512 * 128);
  k_pack_b<<<(1024 * 128) / 256, 256, 0, stream>>>(Wm, WmB, 1024, 128, 1024 * 128);
  k_pack_b<<<(640 * 640) / 256, 256, 0, stream>>>(W1, W1B, 640, 640, 640 * 640);
  k_pack_b<<<(640 * 256) / 256, 256, 0, stream>>>(W2, W2B, 640, 256, 640 * 256);

  // stage 1: QKV projections (24 GEMMs)
  k_qkv<<<dim3(32, 24), 256, 0, stream>>>(xA, WqB, WkB, WvB, bq, bk, bvv, qA, kB, vB);

  // stage 2: fused flash attention (async double-buffered K/V staging)
  k_attn<<<256, 256, 0, stream>>>(qA, kB, vB, mA);

  // stage 3: merge projection
  k_mproj<<<32, 256, 0, stream>>>(mA, WmB, bm, hA);

  // stage 4: layer 1 + BN + relu
  k_l1<<<dim3(32, 5), 256, 0, stream>>>(hA, W1B, b1, Z1);
  k_bn_stats<<<10, 256, 0, stream>>>(Z1, 640, mu1, ri1);
  k_bn_apply_pack<<<(4096 * 640) / 256, 256, 0, stream>>>(Z1, mu1, ri1, g1, be1, 640, 20, h1A);

  // stage 5: layer 2 + BN + relu -> output
  k_l2<<<dim3(32, 2), 256, 0, stream>>>(h1A, W2B, b2, Z2);
  k_bn_stats<<<4, 256, 0, stream>>>(Z2, 256, mu2, ri2);
  k_bn_apply_out<<<(4096 * 256) / 256, 256, 0, stream>>>(Z2, mu2, ri2, g2, be2, 256, out);
}
